// MultiHeadAttention_2327872274601
// MI455X (gfx1250) — matrix operand order repaired and checked by reading, not compiled, not run on hardware
//
#include <hip/hip_runtime.h>

// ---------------------------------------------------------------------------
// MI455X (gfx1250) multi-head attention
//   - all matmuls via v_wmma_f32_16x16x32_f16 (f32 accumulate)
//   - one-time f32->f16 conversion, then async LDS staging (ASYNCcnt path)
// B=4, S=1024, D=1024, H=16, DK=64
// ---------------------------------------------------------------------------

typedef _Float16 v16h __attribute__((ext_vector_type(16)));
typedef _Float16 v8h  __attribute__((ext_vector_type(8)));
typedef _Float16 v4h  __attribute__((ext_vector_type(4)));
typedef float    v8f  __attribute__((ext_vector_type(8)));
typedef float    v4f  __attribute__((ext_vector_type(4)));
typedef int      gv4i __attribute__((vector_size(16)));   // matches builtin param

constexpr int Bn  = 4;
constexpr int Sq  = 1024;
constexpr int Dm  = 1024;
constexpr int Hh  = 16;
constexpr int DKk = 64;

constexpr int M_  = Bn * Sq;  // 4096
constexpr int K_  = Dm;       // 1024
constexpr int N_  = Dm;       // 1024
constexpr int LDA = 40;       // padded halfs per LDS row (80B, bank-spread)

#if __has_builtin(__builtin_amdgcn_global_load_async_to_lds_b128)
#define HAVE_ASYNC_LDS 1
#endif

__device__ __forceinline__ v16h cat8(v8h lo, v8h hi) {
  return __builtin_shufflevector(lo, hi, 0,1,2,3,4,5,6,7,8,9,10,11,12,13,14,15);
}

// 16-byte global -> LDS copy; async (ASYNCcnt) when the builtin exists.
__device__ __forceinline__ void cp16_g2l(const _Float16* g, _Float16* l) {
#ifdef HAVE_ASYNC_LDS
  __builtin_amdgcn_global_load_async_to_lds_b128(
      (__attribute__((address_space(1))) gv4i*)(void*)g,
      (__attribute__((address_space(3))) gv4i*)(void*)l,
      0, 0);
#else
  *(v8h*)l = *(const v8h*)g;
#endif
}

__device__ __forceinline__ void wait_async_cp() {
#ifdef HAVE_ASYNC_LDS
#if __has_builtin(__builtin_amdgcn_s_wait_asynccnt)
  __builtin_amdgcn_s_wait_asynccnt(0);
#else
  asm volatile("s_wait_asynccnt 0" ::: "memory");
#endif
#endif
}

// ---------------------------------------------------------------------------
// Elementwise f32 -> f16 (n is a multiple of 1024: 1M / 4M)
// ---------------------------------------------------------------------------
__global__ __launch_bounds__(256)
void cvt_f16(const float* __restrict__ in, _Float16* __restrict__ out, int n) {
  int i = (blockIdx.x * 256 + threadIdx.x) * 4;
  if (i < n) {
    v4f g = *(const v4f*)(in + i);
    v4h h = {(_Float16)g[0], (_Float16)g[1], (_Float16)g[2], (_Float16)g[3]};
    *(v4h*)(out + i) = h;
  }
}

// ---------------------------------------------------------------------------
// GEMM: out = A @ W^T + bias.  A: [M_,K_] f16, W: [N_,K_] f16 row-major.
// MODE 0: f16 out at [b][h][s][dk]   (Q, K)
// MODE 1: f16 out at [b][h][dk][s]   (V transposed)
// MODE 2: f32 out at [m][n]          (final projection -> d_out)
// 256 threads = 8 waves; block tile 128(M) x 64(N); K tiled by 32.
// ---------------------------------------------------------------------------
template <int MODE>
__global__ __launch_bounds__(256)
void gemm_xwT(const _Float16* __restrict__ A, const _Float16* __restrict__ W,
              const float* __restrict__ bias, void* __restrict__ out) {
  __shared__ _Float16 As[128 * LDA];   // 10240 B
  __shared__ _Float16 Bs[64 * LDA];    //  5120 B

  const int tid  = threadIdx.x;
  const int lane = tid & 31, wid = tid >> 5;
  const int half = lane >> 4, ln = lane & 15;
  const int wm = wid & 3, wn = wid >> 2;          // 4x2 wave grid
  const int m0 = blockIdx.y * 128, n0 = blockIdx.x * 64;

  // staging coordinates: each 16B chunk = 8 halfs; 4 chunks per 32-half row
  const int arow0 = tid >> 2, ach = tid & 3;      // rows 0..63
  const int arow1 = arow0 + 64;                   // rows 64..127

  v8f c[2][2] = {};

  for (int k0 = 0; k0 < K_; k0 += 32) {
    // --- async-stage A tile (128 x 32 halfs = 8KB) and W tile (64 x 32 = 4KB)
    cp16_g2l(A + (size_t)(m0 + arow0) * K_ + k0 + ach * 8, As + arow0 * LDA + ach * 8);
    cp16_g2l(A + (size_t)(m0 + arow1) * K_ + k0 + ach * 8, As + arow1 * LDA + ach * 8);
    cp16_g2l(W + (size_t)(n0 + arow0) * K_ + k0 + ach * 8, Bs + arow0 * LDA + ach * 8);
    wait_async_cp();
    __syncthreads();

    // --- fragments (ISA 16-bit A 16x32 / B 32x16 lane layouts) ---
    v16h a[2], b[2];
#pragma unroll
    for (int i = 0; i < 2; ++i) {
      const _Float16* ap = As + (wm * 32 + i * 16 + ln) * LDA;
      a[i] = cat8(*(const v8h*)(ap + 8 * half), *(const v8h*)(ap + 16 + 8 * half));
      const _Float16* bp = Bs + (wn * 32 + i * 16 + ln) * LDA;
      b[i] = cat8(*(const v8h*)(bp + 8 * half), *(const v8h*)(bp + 16 + 8 * half));
    }
#pragma unroll
    for (int i = 0; i < 2; ++i)
#pragma unroll
      for (int j = 0; j < 2; ++j)
        c[i][j] = __builtin_amdgcn_wmma_f32_16x16x32_f16(
            false, a[i], false, b[j], (short)0, c[i][j], false, false);
    __syncthreads();
  }

  // --- epilogue: bias + layout-specific store (branch-free per MODE) ---
#pragma unroll
  for (int i = 0; i < 2; ++i) {
#pragma unroll
    for (int j = 0; j < 2; ++j) {
      const int n  = n0 + wn * 32 + j * 16 + ln;
      const float bv = bias[n];
#pragma unroll
      for (int r = 0; r < 8; ++r) {
        const int m = m0 + wm * 32 + i * 16 + r + 8 * half;  // C/D: M = r + 8*half
        const float v = c[i][j][r] + bv;
        if (MODE == 2) {
          ((float*)out)[(size_t)m * N_ + n] = v;
        } else {
          const int bb = m >> 10, s = m & 1023;   // m = b*S + s
          const int hh = n >> 6,  dk = n & 63;    // n = h*DK + dk
          size_t off;
          if (MODE == 0) off = ((size_t)(bb * Hh + hh) * Sq + s) * DKk + dk;
          else           off = ((size_t)(bb * Hh + hh) * DKk + dk) * Sq + s;
          ((_Float16*)out)[off] = (_Float16)v;
        }
      }
    }
  }
}

// ---------------------------------------------------------------------------
// Flash attention: one wave per (b, h, 16-query tile); 32-key steps.
// Q,K in [B,H,S,DK] f16; V in [B,H,DK,S] f16. Online softmax via __shfl_xor;
// P transposed through per-wave LDS tile (same-wave in-order LDS + dscnt fence).
// All fragment-element loops fully unrolled (no v_movrels dynamic indexing).
// ---------------------------------------------------------------------------
__global__ __launch_bounds__(128)
void flash_attn(const _Float16* __restrict__ Qh, const _Float16* __restrict__ Kh,
                const _Float16* __restrict__ Vt, const int* __restrict__ mask,
                _Float16* __restrict__ Xh) {
  __shared__ _Float16 plds[4][16 * 48];

  const int tid  = threadIdx.x;
  const int lane = tid & 31, wid = tid >> 5;
  const int half = lane >> 4, ln = lane & 15;
  const int wg = blockIdx.x * 4 + wid;             // 0..4095
  const int qt = wg & 63, hh = (wg >> 6) & 15, bb = wg >> 10;
  const int q0 = qt * 16;

  const _Float16* qbase = Qh + ((size_t)(bb * Hh + hh) * Sq + q0 + ln) * DKk;
  v16h qf[2];
#pragma unroll
  for (int kk = 0; kk < 2; ++kk)
    qf[kk] = cat8(*(const v8h*)(qbase + kk * 32 + 16 * half),
                  *(const v8h*)(qbase + kk * 32 + 16 * half + 8));

  const _Float16* kbase = Kh + (size_t)(bb * Hh + hh) * Sq * DKk;
  const _Float16* vbase = Vt + (size_t)(bb * Hh + hh) * DKk * Sq;
  const int* mbase = mask + (size_t)bb * Sq * Sq;

  float mstat[8], lstat[8];
  v8f o[4] = {};
#pragma unroll
  for (int r = 0; r < 8; ++r) { mstat[r] = -3.0e38f; lstat[r] = 0.0f; }
  _Float16* pl = plds[wid];

  for (int j = 0; j < Sq / 32; ++j) {
    const int key0 = j * 32;

    // S = Q @ K^T : two 16-key column fragments, 2 K-steps each
    v8f s0 = {}, s1 = {};
#pragma unroll
    for (int kk = 0; kk < 2; ++kk) {
      v16h b0 = *(const v16h*)(kbase + (size_t)(key0 + ln) * DKk + kk * 32 + 16 * half);
      v16h b1 = *(const v16h*)(kbase + (size_t)(key0 + 16 + ln) * DKk + kk * 32 + 16 * half);
      s0 = __builtin_amdgcn_wmma_f32_16x16x32_f16(false, qf[kk], false, b0, (short)0, s0, false, false);
      s1 = __builtin_amdgcn_wmma_f32_16x16x32_f16(false, qf[kk], false, b1, (short)0, s1, false, false);
    }

    // scale + mask + online softmax (rows live across 16-lane half-groups)
#pragma unroll
    for (int r = 0; r < 8; ++r) {
      const int qrow = q0 + r + 8 * half;
      const int* mr = mbase + (size_t)qrow * Sq + key0;
      float v0 = (mr[ln]      == 0) ? -65500.0f : s0[r] * 0.125f;
      float v1 = (mr[16 + ln] == 0) ? -65500.0f : s1[r] * 0.125f;
      float mx = fmaxf(v0, v1);
#pragma unroll
      for (int d = 1; d <= 8; d <<= 1) mx = fmaxf(mx, __shfl_xor(mx, d, 32));
      const float mnew = fmaxf(mstat[r], mx);
      const float resc = __expf(mstat[r] - mnew);
      const float p0 = __expf(v0 - mnew);
      const float p1 = __expf(v1 - mnew);
      float rs = p0 + p1;
#pragma unroll
      for (int d = 1; d <= 8; d <<= 1) rs += __shfl_xor(rs, d, 32);
      lstat[r] = lstat[r] * resc + rs;
      mstat[r] = mnew;
#pragma unroll
      for (int t = 0; t < 4; ++t) o[t][r] *= resc;
      pl[(r + 8 * half) * 48 + ln]      = (_Float16)p0;
      pl[(r + 8 * half) * 48 + 16 + ln] = (_Float16)p1;
    }
    asm volatile("s_wait_dscnt 0" ::: "memory");   // same-wave LDS RAW fence

    // O += P @ V
    const _Float16* pr = pl + ln * 48;
    v16h pa = cat8(*(const v8h*)(pr + 16 * half), *(const v8h*)(pr + 16 * half + 8));
#pragma unroll
    for (int t = 0; t < 4; ++t) {
      v16h vb = *(const v16h*)(vbase + (size_t)(t * 16 + ln) * Sq + key0 + 16 * half);
      o[t] = __builtin_amdgcn_wmma_f32_16x16x32_f16(false, pa, false, vb, (short)0, o[t], false, false);
    }
  }

#pragma unroll
  for (int r = 0; r < 8; ++r) {
    const int qrow = q0 + r + 8 * half;
    const float inv = 1.0f / lstat[r];
    _Float16* xr = Xh + (size_t)(bb * Sq + qrow) * Dm + hh * DKk;
#pragma unroll
    for (int t = 0; t < 4; ++t)
      xr[t * 16 + ln] = (_Float16)(o[t][r] * inv);
  }
}

// ---------------------------------------------------------------------------
extern "C" void kernel_launch(void* const* d_in, const int* in_sizes, int n_in,
                              void* d_out, int out_size, void* d_ws, size_t ws_size,
                              hipStream_t stream) {
  (void)in_sizes; (void)n_in; (void)out_size; (void)ws_size;

  const float* query  = (const float*)d_in[0];
  const float* key_in = (const float*)d_in[1];
  const float* value  = (const float*)d_in[2];
  const int*   mask   = (const int*)  d_in[3];
  const float* Wq     = (const float*)d_in[4];
  const float* bq     = (const float*)d_in[5];
  const float* Wk     = (const float*)d_in[6];
  const float* bk     = (const float*)d_in[7];
  const float* Wv     = (const float*)d_in[8];
  const float* bv     = (const float*)d_in[9];
  const float* Wo     = (const float*)d_in[10];
  const float* bo     = (const float*)d_in[11];

  constexpr size_t MB = 1024 * 1024;
  char* ws = (char*)d_ws;
  _Float16* Aq  = (_Float16*)(ws);            //  8MB, reused as Xh later
  _Float16* Ak  = (_Float16*)(ws +  8 * MB);  //  8MB
  _Float16* Av  = (_Float16*)(ws + 16 * MB);  //  8MB
  _Float16* Wqh = (_Float16*)(ws + 24 * MB);  //  2MB
  _Float16* Wkh = (_Float16*)(ws + 26 * MB);  //  2MB
  _Float16* Wvh = (_Float16*)(ws + 28 * MB);  //  2MB
  _Float16* Woh = (_Float16*)(ws + 30 * MB);  //  2MB
  _Float16* Qh  = (_Float16*)(ws + 32 * MB);  //  8MB
  _Float16* Kh  = (_Float16*)(ws + 40 * MB);  //  8MB
  _Float16* Vt  = (_Float16*)(ws + 48 * MB);  //  8MB  (total 56MB)
  _Float16* Xh  = Aq;                         //  aliases dead Aq

  const int nAct = Bn * Sq * Dm;   // 4M
  const int nWgt = Dm * Dm;        // 1M
  cvt_f16<<<nAct / 1024, 256, 0, stream>>>(query,  Aq,  nAct);
  cvt_f16<<<nAct / 1024, 256, 0, stream>>>(key_in, Ak,  nAct);
  cvt_f16<<<nAct / 1024, 256, 0, stream>>>(value,  Av,  nAct);
  cvt_f16<<<nWgt / 1024, 256, 0, stream>>>(Wq,     Wqh, nWgt);
  cvt_f16<<<nWgt / 1024, 256, 0, stream>>>(Wk,     Wkh, nWgt);
  cvt_f16<<<nWgt / 1024, 256, 0, stream>>>(Wv,     Wvh, nWgt);
  cvt_f16<<<nWgt / 1024, 256, 0, stream>>>(Wo,     Woh, nWgt);

  dim3 gGrid(N_ / 64, M_ / 128);   // (16, 32)
  dim3 gBlk(256);
  gemm_xwT<0><<<gGrid, gBlk, 0, stream>>>(Aq, Wqh, bq, Qh);
  gemm_xwT<0><<<gGrid, gBlk, 0, stream>>>(Ak, Wkh, bk, Kh);
  gemm_xwT<1><<<gGrid, gBlk, 0, stream>>>(Av, Wvh, bv, Vt);

  flash_attn<<<Bn * Hh * (Sq / 16) / 4, 128, 0, stream>>>(Qh, Kh, Vt, mask, Xh);

  gemm_xwT<2><<<gGrid, gBlk, 0, stream>>>(Xh, Woh, bo, d_out);
}
